// BLSTMModel_24137716203605
// MI455X (gfx1250) — matrix operand order repaired and checked by reading, not compiled, not run on hardware
//
#include <hip/hip_runtime.h>
#include <hip/hip_bf16.h>

// ---------------------------------------------------------------------------
// BLSTM (B=16,S=128,V=32000,D=256,H=256) for MI455X / gfx1250 (wave32, WMMA)
// Pipeline:
//   1) f32->f16 weight conversions (w_ih_f, w_ih_b, w_hh_f, fc_w)
//   2) embedding gather -> f16
//   3) WMMA GEMM: xp_f = emb @ w_ih_f^T + b_ih_f + b_hh_f   (f32 out)
//                 gb   = emb @ w_ih_b^T + b_ih_b + b_hh_b   (f32 out)
//   4) backward single-cell elementwise -> hb (f16, cols 256..511 of outcat)
//   5) persistent single-WG LSTM scan (WMMA h@w_hh^T per step, state in LDS)
//   6) WMMA GEMM: logits = outcat @ fc_w^T + fc_b
//      - 2 M-tiles per wave (B-fragment reuse), A staged via async-to-LDS,
//        non-temporal stores for the 262MB streaming output.
// All matmuls use V_WMMA_F32_16X16X32_F16 (f16 in, f32 accumulate).
// ---------------------------------------------------------------------------

typedef _Float16 v16h __attribute__((ext_vector_type(16)));
typedef _Float16 v8h  __attribute__((ext_vector_type(8)));
typedef float    v8f  __attribute__((ext_vector_type(8)));
typedef int      v4i  __attribute__((ext_vector_type(4)));

#define BB   16
#define SS   128
#define VV   32000
#define DD   256
#define HH   256
#define G4   1024   // 4*H
#define K2   512    // 2*H
#define NROW (BB*SS) // 2048

// ---- workspace layout (bytes) ----
#define OFF_EMB   (0)                                   // f16 [2048][256]   1 MB
#define OFF_WIHF  (OFF_EMB  + 2048*256*2)               // f16 [1024][256] 512 KB
#define OFF_WIHB  (OFF_WIHF + 1024*256*2)               // f16 [1024][256] 512 KB
#define OFF_WHH   (OFF_WIHB + 1024*256*2)               // f16 [1024][256] 512 KB
#define OFF_FCW   (OFF_WHH  + 1024*256*2)               // f16 [32000][512] 32 MB
#define OFF_XPF   (OFF_FCW  + 32000*512*2)              // f32 [2048][1024]  8 MB
#define OFF_GB    (OFF_XPF  + 2048*1024*4)              // f32 [2048][1024]  8 MB
#define OFF_OUT   (OFF_GB   + 2048*1024*4)              // f16 [2048][512]   2 MB

// ---- gfx1250 async-to-LDS path (guarded; falls back to sync staging) ----
#if defined(__has_builtin)
#if __has_builtin(__builtin_amdgcn_global_load_async_to_lds_b128)
#define HAVE_ASYNC_LDS 1
#endif
#endif

#ifdef HAVE_ASYNC_LDS
// Prototype (from hipcc diagnostic): param0 = v4i in addrspace(1) [global src],
// param1 = v4i in addrspace(3) [LDS dst], then imm offset, imm cpol.
typedef __attribute__((address_space(1))) v4i as1_v4i;
typedef __attribute__((address_space(3))) v4i as3_v4i;
#endif

static __device__ __forceinline__ void wait_async0() {
#if defined(__has_builtin) && __has_builtin(__builtin_amdgcn_s_wait_asynccnt)
    __builtin_amdgcn_s_wait_asynccnt(0);
#else
    asm volatile("s_wait_asynccnt 0" ::: "memory");
#endif
}

static __device__ __forceinline__ float sigf(float x) {
    return 1.0f / (1.0f + __expf(-x));
}

// A fragment: 16-bit A 16x32 (MxK). Row pointer is per-lane (row = lane&15).
// elements 0..7  = A[row][kk + 8*half .. +7]
// elements 8..15 = A[row][kk + 16 + 8*half .. +7]
static __device__ __forceinline__ v16h frag_A(const _Float16* __restrict__ rowp,
                                              int kk, int lane) {
    int hh = (lane >> 4) & 1;
    v8h lo = *(const v8h*)(rowp + kk + 8 * hh);
    v8h hi = *(const v8h*)(rowp + kk + 16 + 8 * hh);
    v16h r;
#pragma unroll
    for (int i = 0; i < 8; ++i) { r[i] = lo[i]; r[8 + i] = hi[i]; }
    return r;
}

// B fragment: 16-bit B 32x16 (KxN) sourced from row-major W[N][K] (=> B = W^T).
// elements 0..7 = K=kk+8*half..+7, elements 8..15 = K=kk+16+8*half..+7 (N=lane&15).
static __device__ __forceinline__ v16h frag_B(const _Float16* __restrict__ rowp,
                                              int kk, int lane) {
    int hh = (lane >> 4) & 1;
    const _Float16* p = rowp + kk + 8 * hh;
    v8h lo = *(const v8h*)(p);
    v8h hi = *(const v8h*)(p + 16);
    v16h r;
#pragma unroll
    for (int i = 0; i < 8; ++i) { r[i] = lo[i]; r[8 + i] = hi[i]; }
    return r;
}

// ---------------------------------------------------------------------------
// 1) f32 -> f16 conversion
__global__ __launch_bounds__(256) void cvt_h_kernel(const float* __restrict__ src,
                                                    _Float16* __restrict__ dst, int n) {
    int i = blockIdx.x * 256 + threadIdx.x;
    if (i < n) dst[i] = (_Float16)src[i];
}

// ---------------------------------------------------------------------------
// 2) embedding gather (f32 table -> f16 rows)
__global__ __launch_bounds__(256) void gather_kernel(const int* __restrict__ x,
                                                     const float* __restrict__ embed,
                                                     _Float16* __restrict__ emb_h) {
    int row = blockIdx.x;          // 0..2047 (= b*S + s)
    int d   = threadIdx.x;         // 0..255
    int tok = x[row];
    emb_h[(size_t)row * DD + d] = (_Float16)embed[(size_t)tok * DD + d];
}

// ---------------------------------------------------------------------------
// 3) input-projection GEMM: out[2048][1024] = A[2048][256] * W[1024][256]^T + b0 + b1
//    8 waves/block, one 16x16 tile each. grid = (64 Ntiles, 16 Mgroups)
__global__ __launch_bounds__(256) void gemm_xp_kernel(const _Float16* __restrict__ A,
                                                      const _Float16* __restrict__ W,
                                                      const float* __restrict__ b0,
                                                      const float* __restrict__ b1,
                                                      float* __restrict__ out) {
    int lane = threadIdx.x & 31, wave = threadIdx.x >> 5;
    int mt = blockIdx.y * 8 + wave;        // 0..127
    int nt = blockIdx.x;                   // 0..63
    int n  = lane & 15, hh = lane >> 4;
    int col  = nt * 16 + n;
    int rowA = mt * 16 + (lane & 15);
    const _Float16* arow = A + (size_t)rowA * DD;
    const _Float16* brow = W + (size_t)col * DD;
    v8f acc = {};
#pragma unroll
    for (int kk = 0; kk < DD; kk += 32) {
        v16h a = frag_A(arow, kk, lane);
        v16h b = frag_B(brow, kk, lane);
        acc = __builtin_amdgcn_wmma_f32_16x16x32_f16(false, a, false, b,
                                                     (short)0, acc, false, false);
    }
    float bias = b0[col] + b1[col];
#pragma unroll
    for (int r = 0; r < 8; ++r) {
        int m = r + 8 * hh;
        out[(size_t)(mt * 16 + m) * G4 + col] = acc[r] + bias;
    }
}

// ---------------------------------------------------------------------------
// 4) backward direction = single LSTM cell step from zero state (f gate moot)
__global__ __launch_bounds__(256) void bwd_cell_kernel(const float* __restrict__ g,
                                                       _Float16* __restrict__ outcat) {
    int idx = blockIdx.x * 256 + threadIdx.x;     // 2048*256
    if (idx >= NROW * HH) return;
    int row = idx >> 8, j = idx & 255;
    const float* gr = g + (size_t)row * G4;
    float cb = sigf(gr[j]) * tanhf(gr[512 + j]);
    float hb = sigf(gr[768 + j]) * tanhf(cb);
    outcat[(size_t)row * K2 + 256 + j] = (_Float16)hb;
}

// ---------------------------------------------------------------------------
// 5) persistent LSTM scan: one workgroup, 16 waves, 128 sequential steps.
//    Wave w owns N-tiles [4w, 4w+4) of the 4H=1024 gate columns; M=16 = batch.
//    State h (f16) / c (f32) live in LDS; gates staged in LDS (f16). 56 KB LDS.
__global__ __launch_bounds__(512) void lstm_scan_kernel(const float* __restrict__ xp,
                                                        const _Float16* __restrict__ whh,
                                                        _Float16* __restrict__ outcat) {
    __shared__ __align__(16) _Float16 h_sh[16 * 256];
    __shared__ __align__(16) float    c_sh[16 * 256];
    __shared__ __align__(16) _Float16 g_sh[16 * 1024];
    int tid  = threadIdx.x;
    int lane = tid & 31, wave = tid >> 5;
    for (int i = tid; i < 16 * 256; i += 512) {
        h_sh[i] = (_Float16)0.0f;
        c_sh[i] = 0.0f;
    }
    __syncthreads();
    int n = lane & 15, hh = lane >> 4;
    for (int t = 0; t < SS; ++t) {
        // gates = xp[:,t,:] + h @ whh^T   (4 tiles per wave)
#pragma unroll
        for (int tile = 0; tile < 4; ++tile) {
            int nt  = wave * 4 + tile;        // 0..63
            int col = nt * 16 + n;
            v8f acc;
#pragma unroll
            for (int r = 0; r < 8; ++r) {
                int m = r + 8 * hh;           // batch index
                acc[r] = xp[(size_t)(m * SS + t) * G4 + col];
            }
            const _Float16* arow = h_sh + (lane & 15) * 256;
            const _Float16* brow = whh + (size_t)col * DD;
#pragma unroll
            for (int kk = 0; kk < DD; kk += 32) {
                v16h a = frag_A(arow, kk, lane);
                v16h b = frag_B(brow, kk, lane);
                acc = __builtin_amdgcn_wmma_f32_16x16x32_f16(false, a, false, b,
                                                             (short)0, acc, false, false);
            }
#pragma unroll
            for (int r = 0; r < 8; ++r) {
                int m = r + 8 * hh;
                g_sh[m * 1024 + col] = (_Float16)acc[r];
            }
        }
        __syncthreads();
        // elementwise cell update: 16x256 elements
        for (int idx = tid; idx < 16 * HH; idx += 512) {
            int m = idx >> 8, j = idx & 255;
            float ig = sigf((float)g_sh[m * 1024 + j]);
            float fg = sigf((float)g_sh[m * 1024 + 256 + j]);
            float gg = tanhf((float)g_sh[m * 1024 + 512 + j]);
            float og = sigf((float)g_sh[m * 1024 + 768 + j]);
            float cc = fg * c_sh[m * 256 + j] + ig * gg;
            float hv = og * tanhf(cc);
            c_sh[m * 256 + j] = cc;
            h_sh[m * 256 + j] = (_Float16)hv;
            outcat[(size_t)(m * SS + t) * K2 + j] = (_Float16)hv;
        }
        __syncthreads();
    }
}

// ---------------------------------------------------------------------------
// 6) classifier GEMM: logits[2048][32000] = A[2048][512] * fcW[32000][512]^T + fc_b
//    Block: 8 waves, each wave owns one N-tile and TWO M-tiles (B-frag reuse).
//    A (32 rows x 512 f16 = 32KB) staged via async-to-LDS; D staged through LDS
//    (16KB) for coalesced non-temporal row writes. grid = (250, 64).
__global__ __launch_bounds__(256) void gemm_fc_kernel(const _Float16* __restrict__ A,
                                                      const _Float16* __restrict__ W,
                                                      const float* __restrict__ bias,
                                                      float* __restrict__ out) {
    __shared__ __align__(16) _Float16 a_sh[32 * K2];   // 32 KB
    __shared__ __align__(16) float    d_sh[32 * 128];  // 16 KB
    int tid = threadIdx.x, lane = tid & 31, wave = tid >> 5;
    int mt = blockIdx.y;                   // 0..63  (32 rows each)
    int ncolBase = blockIdx.x * 128;       // 0..249 * 128
    // stage A tile (32 rows x 512 halves) cooperatively, 128-bit chunks
    {
        const v4i* asrc = (const v4i*)(A + (size_t)mt * 32 * K2);
        v4i* adst = (v4i*)a_sh;
#ifdef HAVE_ASYNC_LDS
        for (int i = tid; i < 32 * K2 / 8; i += 256) {
            __builtin_amdgcn_global_load_async_to_lds_b128(
                (as1_v4i*)(asrc + i), (as3_v4i*)(adst + i), 0, 0);
        }
        wait_async0();
#else
        for (int i = tid; i < 32 * K2 / 8; i += 256) adst[i] = asrc[i];
#endif
    }
    __syncthreads();
    int n = lane & 15, hh = lane >> 4;
    int col = ncolBase + wave * 16 + n;
    const _Float16* arow0 = a_sh + (lane & 15) * K2;        // M-tile 0 (rows 0..15)
    const _Float16* arow1 = arow0 + 16 * K2;                // M-tile 1 (rows 16..31)
    const _Float16* brow  = W + (size_t)col * K2;
    __builtin_prefetch(brow, 0, 1);   // global_prefetch_b8 for the weight row
    v8f acc0 = {}, acc1 = {};
#pragma unroll
    for (int kk = 0; kk < K2; kk += 32) {
        v16h b  = frag_B(brow, kk, lane);                   // loaded once,
        v16h a0 = frag_A(arow0, kk, lane);                  // used twice
        v16h a1 = frag_A(arow1, kk, lane);
        acc0 = __builtin_amdgcn_wmma_f32_16x16x32_f16(false, a0, false, b,
                                                      (short)0, acc0, false, false);
        acc1 = __builtin_amdgcn_wmma_f32_16x16x32_f16(false, a1, false, b,
                                                      (short)0, acc1, false, false);
    }
#pragma unroll
    for (int r = 0; r < 8; ++r) {
        int m = r + 8 * hh;
        d_sh[m * 128 + wave * 16 + n]        = acc0[r];
        d_sh[(16 + m) * 128 + wave * 16 + n] = acc1[r];
    }
    __syncthreads();
    // coalesced streaming writes; NT hint keeps fc_w resident in L2
    for (int i = tid; i < 32 * 128; i += 256) {
        int m = i >> 7, c = i & 127;
        int gcol = ncolBase + c;
        float v = d_sh[i] + bias[gcol];
        __builtin_nontemporal_store(v, &out[(size_t)(mt * 32 + m) * VV + gcol]);
    }
}

// ---------------------------------------------------------------------------
extern "C" void kernel_launch(void* const* d_in, const int* in_sizes, int n_in,
                              void* d_out, int out_size, void* d_ws, size_t ws_size,
                              hipStream_t stream) {
    (void)in_sizes; (void)n_in; (void)out_size; (void)ws_size;
    const int*   x      = (const int*)d_in[0];
    const float* embed  = (const float*)d_in[1];
    const float* w_ih_f = (const float*)d_in[2];
    const float* w_hh_f = (const float*)d_in[3];
    const float* b_ih_f = (const float*)d_in[4];
    const float* b_hh_f = (const float*)d_in[5];
    const float* w_ih_b = (const float*)d_in[6];
    /* d_in[7] = w_hh_b: unused (backward scan starts from zero state) */
    const float* b_ih_b = (const float*)d_in[8];
    const float* b_hh_b = (const float*)d_in[9];
    const float* fc_w   = (const float*)d_in[10];
    const float* fc_b   = (const float*)d_in[11];
    float* logits = (float*)d_out;

    char* ws = (char*)d_ws;
    _Float16* emb_h  = (_Float16*)(ws + OFF_EMB);
    _Float16* wihf_h = (_Float16*)(ws + OFF_WIHF);
    _Float16* wihb_h = (_Float16*)(ws + OFF_WIHB);
    _Float16* whh_h  = (_Float16*)(ws + OFF_WHH);
    _Float16* fcw_h  = (_Float16*)(ws + OFF_FCW);
    float*    xpf    = (float*)(ws + OFF_XPF);
    float*    gb     = (float*)(ws + OFF_GB);
    _Float16* outcat = (_Float16*)(ws + OFF_OUT);

    // 1) weight conversions
    {
        int n = G4 * DD; // 262144
        cvt_h_kernel<<<(n + 255) / 256, 256, 0, stream>>>(w_ih_f, wihf_h, n);
        cvt_h_kernel<<<(n + 255) / 256, 256, 0, stream>>>(w_ih_b, wihb_h, n);
        cvt_h_kernel<<<(n + 255) / 256, 256, 0, stream>>>(w_hh_f, whh_h, n);
        int nf = VV * K2; // 16,384,000
        cvt_h_kernel<<<(nf + 255) / 256, 256, 0, stream>>>(fc_w, fcw_h, nf);
    }
    // 2) embedding gather
    gather_kernel<<<NROW, 256, 0, stream>>>(x, embed, emb_h);
    // 3) input projections (fwd gates incl. both biases; bwd gates likewise)
    gemm_xp_kernel<<<dim3(64, 16), 256, 0, stream>>>(emb_h, wihf_h, b_ih_f, b_hh_f, xpf);
    gemm_xp_kernel<<<dim3(64, 16), 256, 0, stream>>>(emb_h, wihb_h, b_ih_b, b_hh_b, gb);
    // 4) backward single-step cell -> hb half of outcat
    bwd_cell_kernel<<<(NROW * HH + 255) / 256, 256, 0, stream>>>(gb, outcat);
    // 5) forward recurrence (persistent single workgroup) -> hf half of outcat
    lstm_scan_kernel<<<1, 512, 0, stream>>>(xpf, whh_h, outcat);
    // 6) classifier GEMM -> logits
    gemm_fc_kernel<<<dim3(250, 64), 256, 0, stream>>>(outcat, fcw_h, fc_b, logits);
}
